// SAGPool_58033598104009
// MI455X (gfx1250) — matrix operand order repaired and checked by reading, not compiled, not run on hardware
//
#include <hip/hip_runtime.h>
#include <hip/hip_bf16.h>

typedef __bf16 bf16_t;
typedef __attribute__((ext_vector_type(16))) __bf16 v16bf;
typedef __attribute__((ext_vector_type(8)))  float  v8f;

static constexpr int cG    = 32;
static constexpr int cNPER = 1024;
static constexpr int cN    = 32768;     // G*N_PER
static constexpr int cE    = 524288;    // N*DEG
static constexpr int cHID  = 128;
static constexpr int cK    = 820;       // ceil(0.8*1024)
static constexpr int cN2   = 26240;     // G*K
static constexpr int cNCLS = 10;

// ---------------- device helpers ----------------

__device__ __forceinline__ float eluf(float v) {
    return v > 0.0f ? v : expm1f(v);
}

// float -> bf16; native cast lowers to v_cvt_pk_bf16_f32-class ops on gfx1250
__device__ __forceinline__ bf16_t tobf(float f) {
    return (bf16_t)f;
}

// ---------------- tiny utility kernels ----------------

__global__ void zerof_kernel(float* p, int n) {
    int i = blockIdx.x * blockDim.x + threadIdx.x;
    if (i < n) p[i] = 0.0f;
}

__global__ void filli_kernel(int* p, int n, int val) {
    int i = blockIdx.x * blockDim.x + threadIdx.x;
    if (i < n) p[i] = val;
}

// Pack row-major W[k][n] (K = KT*32, N = 128) into WMMA B-fragment order (bf16):
// Wp[((nt*KT + t)*32 + lane)*16 + j] = W[(t*32 + 8*(lane/16) + (j<8 ? j : j+8))*128 + nt*16 + lane%16]
__global__ void pack_w_kernel(const float* __restrict__ W, bf16_t* __restrict__ Wp, int KT) {
    int idx = blockIdx.x * blockDim.x + threadIdx.x;
    int total = 8 * KT * 512;
    if (idx >= total) return;
    int j    = idx & 15;
    int lane = (idx >> 4) & 31;
    int grp  = idx >> 9;          // nt*KT + t
    int t    = grp % KT;
    int nt   = grp / KT;
    int k = t * 32 + 8 * (lane >> 4) + ((j < 8) ? j : (j + 8));
    int n = nt * 16 + (lane & 15);
    Wp[idx] = tobf(W[k * 128 + n]);
}

// stats[2i]=S, stats[2i+1]=Q  ->  stats[16+2i]=mean, stats[17+2i]=rsqrt(var)
__global__ void finalize_stats_kernel(float* stats, int idx, float totFixed, const int* vcnt) {
    if (threadIdx.x == 0 && blockIdx.x == 0) {
        float tot  = vcnt ? ((float)(*vcnt) * 128.0f) : totFixed;
        float S    = stats[2 * idx];
        float Q    = stats[2 * idx + 1];
        float mean = S / tot;
        float var  = Q / tot - mean * mean;
        stats[16 + 2 * idx] = mean;
        stats[17 + 2 * idx] = rsqrtf(fmaxf(var, 1e-20f));
    }
}

// ---------------- EdgeConv layer 1 ----------------

// Pass A: per-edge y1 = elu(feat6 @ W1a + b1a); accumulate global sum / sumsq.
__global__ void __launch_bounds__(256)
conv1_statsA_kernel(const float* __restrict__ x, const int* __restrict__ sidx,
                    const int* __restrict__ didx, const float* __restrict__ W1a,
                    const float* __restrict__ b1a, float* stats) {
    __shared__ float sW[768];
    __shared__ float sB[128];
    __shared__ float rS[256], rQ[256];
    int tid = threadIdx.x;
    for (int i = tid; i < 768; i += 256) sW[i] = W1a[i];
    if (tid < 128) sB[tid] = b1a[tid];
    __syncthreads();

    int e = blockIdx.x * 256 + tid;
    int s = sidx[e], d = didx[e];
    float f0 = x[d * 3 + 0], f1 = x[d * 3 + 1], f2 = x[d * 3 + 2];
    float f3 = x[s * 3 + 0] - f0, f4 = x[s * 3 + 1] - f1, f5 = x[s * 3 + 2] - f2;

    float S = 0.0f, Q = 0.0f;
    for (int k = 0; k < 128; ++k) {
        float y = sB[k] + f0 * sW[k] + f1 * sW[128 + k] + f2 * sW[256 + k]
                        + f3 * sW[384 + k] + f4 * sW[512 + k] + f5 * sW[640 + k];
        y = eluf(y);
        S += y; Q += y * y;
    }
    rS[tid] = S; rQ[tid] = Q;
    __syncthreads();
    for (int st = 128; st > 0; st >>= 1) {
        if (tid < st) { rS[tid] += rS[tid + st]; rQ[tid] += rQ[tid + st]; }
        __syncthreads();
    }
    if (tid == 0) { atomicAdd(&stats[0], rS[0]); atomicAdd(&stats[1], rQ[0]); }
}

// Layer-b WMMA kernel. A = LN(elu(feat6@W1a+b1a)) recomputed on the fly (K=128).
// mode 0: accumulate stats of elu(A@W1b+b1b) into stats[2],stats[3]
// mode 1: normalize with stats[18],[19] and atomic-scatter into outacc[dst].
__global__ void __launch_bounds__(256)
conv1_wmma_kernel(const float* __restrict__ x, const int* __restrict__ sidx,
                  const int* __restrict__ didx, const float* __restrict__ W1a,
                  const float* __restrict__ b1a, const bf16_t* __restrict__ Wp,
                  const float* __restrict__ b1b, float* stats, int mode,
                  float* outacc) {
    __shared__ float sW[768];
    __shared__ float sB[128];
    __shared__ float rS[256], rQ[256];
    int tid = threadIdx.x;
    for (int i = tid; i < 768; i += 256) sW[i] = W1a[i];
    if (tid < 128) sB[tid] = b1a[tid];
    __syncthreads();

    int wave = tid >> 5, lane = tid & 31;
    int hf = lane >> 4, r = lane & 15;
    int ebase = (blockIdx.x * 8 + wave) * 16;
    int e = ebase + r;
    int s = sidx[e], d = didx[e];
    float f0 = x[d * 3 + 0], f1 = x[d * 3 + 1], f2 = x[d * 3 + 2];
    float f3 = x[s * 3 + 0] - f0, f4 = x[s * 3 + 1] - f1, f5 = x[s * 3 + 2] - f2;

    float mean1 = stats[16], inv1 = stats[17];

    v16bf a[4];
#pragma unroll
    for (int t = 0; t < 4; ++t) {
#pragma unroll
        for (int j = 0; j < 16; ++j) {
            int kk = t * 32 + 8 * hf + ((j < 8) ? j : (j + 8));
            float y = sB[kk] + f0 * sW[kk] + f1 * sW[128 + kk] + f2 * sW[256 + kk]
                             + f3 * sW[384 + kk] + f4 * sW[512 + kk] + f5 * sW[640 + kk];
            y = eluf(y);
            a[t][j] = tobf((y - mean1) * inv1);
        }
    }

    v8f zero = {0.f, 0.f, 0.f, 0.f, 0.f, 0.f, 0.f, 0.f};
    v8f acc[8];
#pragma unroll
    for (int i = 0; i < 8; ++i) acc[i] = zero;

#pragma unroll
    for (int nt = 0; nt < 8; ++nt) {
#pragma unroll
        for (int t = 0; t < 4; ++t) {
            v16bf b = *(const v16bf*)(Wp + (((nt * 4 + t) * 32 + lane) << 4));
            acc[nt] = __builtin_amdgcn_wmma_f32_16x16x32_bf16(
                false, a[t], false, b, (short)0, acc[nt], false, false);
        }
    }

    int col0 = lane & 15;
    int mrow = 8 * hf;
    if (mode == 0) {
        float S = 0.0f, Q = 0.0f;
#pragma unroll
        for (int nt = 0; nt < 8; ++nt) {
            float bc = b1b[nt * 16 + col0];
#pragma unroll
            for (int v2 = 0; v2 < 8; ++v2) {
                float y = eluf(acc[nt][v2] + bc);
                S += y; Q += y * y;
            }
        }
        rS[tid] = S; rQ[tid] = Q;
        __syncthreads();
        for (int st = 128; st > 0; st >>= 1) {
            if (tid < st) { rS[tid] += rS[tid + st]; rQ[tid] += rQ[tid + st]; }
            __syncthreads();
        }
        if (tid == 0) { atomicAdd(&stats[2], rS[0]); atomicAdd(&stats[3], rQ[0]); }
    } else {
        float mean2 = stats[18], inv2 = stats[19];
        int dn[8];
#pragma unroll
        for (int v2 = 0; v2 < 8; ++v2) dn[v2] = didx[ebase + mrow + v2];
#pragma unroll
        for (int nt = 0; nt < 8; ++nt) {
            int col = nt * 16 + col0;
            float bc = b1b[col];
#pragma unroll
            for (int v2 = 0; v2 < 8; ++v2) {
                float y = eluf(acc[nt][v2] + bc);
                atomicAdd(&outacc[(size_t)dn[v2] * 128 + col], (y - mean2) * inv2);
            }
        }
    }
}

// ---------------- EdgeConv 2/3 shared WMMA kernel (K=256) ----------------
// A = concat(feat[dst], feat[src]-feat[dst]) (256), W packed bf16 (8 n-tiles x 8 k-tiles).
// mode 0: accumulate masked stats into stats[2*lnIdx .. +1]
// mode 1: normalize with stats[16+2*lnIdx], scatter into outacc[dscat] (masked).
__global__ void __launch_bounds__(256)
edgeconv_wmma256_kernel(const float* __restrict__ feat, const int* __restrict__ sidx,
                        const int* __restrict__ didx, const float* __restrict__ emask,
                        const bf16_t* __restrict__ Wp, const float* __restrict__ bias,
                        float* stats, int lnIdx, int mode, float* outacc,
                        const int* __restrict__ dscat) {
    __shared__ float rS[256], rQ[256];
    int tid = threadIdx.x;
    int wave = tid >> 5, lane = tid & 31;
    int hf = lane >> 4, r = lane & 15;
    int ebase = (blockIdx.x * 8 + wave) * 16;
    int e = ebase + r;
    int s = sidx[e], d = didx[e];
    const float* rowD = feat + (size_t)d * 128;
    const float* rowS = feat + (size_t)s * 128;

    v16bf a[8];
#pragma unroll
    for (int t = 0; t < 8; ++t) {
        int kb = t * 32 + 8 * hf;
        float v[16];
        if (t < 4) {
            float4 u0 = *(const float4*)(rowD + kb);
            float4 u1 = *(const float4*)(rowD + kb + 4);
            float4 u2 = *(const float4*)(rowD + kb + 16);
            float4 u3 = *(const float4*)(rowD + kb + 20);
            v[0] = u0.x;  v[1] = u0.y;  v[2] = u0.z;  v[3] = u0.w;
            v[4] = u1.x;  v[5] = u1.y;  v[6] = u1.z;  v[7] = u1.w;
            v[8] = u2.x;  v[9] = u2.y;  v[10] = u2.z; v[11] = u2.w;
            v[12] = u3.x; v[13] = u3.y; v[14] = u3.z; v[15] = u3.w;
        } else {
            int kc = kb - 128;
            float4 a0 = *(const float4*)(rowS + kc);
            float4 b0 = *(const float4*)(rowD + kc);
            float4 a1 = *(const float4*)(rowS + kc + 4);
            float4 b1 = *(const float4*)(rowD + kc + 4);
            float4 a2 = *(const float4*)(rowS + kc + 16);
            float4 b2 = *(const float4*)(rowD + kc + 16);
            float4 a3 = *(const float4*)(rowS + kc + 20);
            float4 b3 = *(const float4*)(rowD + kc + 20);
            v[0] = a0.x - b0.x;  v[1] = a0.y - b0.y;  v[2] = a0.z - b0.z;  v[3] = a0.w - b0.w;
            v[4] = a1.x - b1.x;  v[5] = a1.y - b1.y;  v[6] = a1.z - b1.z;  v[7] = a1.w - b1.w;
            v[8] = a2.x - b2.x;  v[9] = a2.y - b2.y;  v[10] = a2.z - b2.z; v[11] = a2.w - b2.w;
            v[12] = a3.x - b3.x; v[13] = a3.y - b3.y; v[14] = a3.z - b3.z; v[15] = a3.w - b3.w;
        }
#pragma unroll
        for (int j = 0; j < 16; ++j) a[t][j] = tobf(v[j]);
    }

    v8f zero = {0.f, 0.f, 0.f, 0.f, 0.f, 0.f, 0.f, 0.f};
    v8f acc[8];
#pragma unroll
    for (int i = 0; i < 8; ++i) acc[i] = zero;

#pragma unroll
    for (int nt = 0; nt < 8; ++nt) {
#pragma unroll
        for (int t = 0; t < 8; ++t) {
            v16bf b = *(const v16bf*)(Wp + (((nt * 8 + t) * 32 + lane) << 4));
            acc[nt] = __builtin_amdgcn_wmma_f32_16x16x32_bf16(
                false, a[t], false, b, (short)0, acc[nt], false, false);
        }
    }

    int col0 = lane & 15;
    int mrow = 8 * hf;
    float mks[8];
#pragma unroll
    for (int v2 = 0; v2 < 8; ++v2)
        mks[v2] = emask ? emask[ebase + mrow + v2] : 1.0f;

    if (mode == 0) {
        float S = 0.0f, Q = 0.0f;
#pragma unroll
        for (int nt = 0; nt < 8; ++nt) {
            float bc = bias[nt * 16 + col0];
#pragma unroll
            for (int v2 = 0; v2 < 8; ++v2) {
                float y = eluf(acc[nt][v2] + bc);
                S += y * mks[v2];
                Q += y * y * mks[v2];
            }
        }
        rS[tid] = S; rQ[tid] = Q;
        __syncthreads();
        for (int st = 128; st > 0; st >>= 1) {
            if (tid < st) { rS[tid] += rS[tid + st]; rQ[tid] += rQ[tid + st]; }
            __syncthreads();
        }
        if (tid == 0) {
            atomicAdd(&stats[2 * lnIdx], rS[0]);
            atomicAdd(&stats[2 * lnIdx + 1], rQ[0]);
        }
    } else {
        float mean = stats[16 + 2 * lnIdx], inv = stats[17 + 2 * lnIdx];
        int dn[8];
#pragma unroll
        for (int v2 = 0; v2 < 8; ++v2) dn[v2] = dscat[ebase + mrow + v2];
#pragma unroll
        for (int nt = 0; nt < 8; ++nt) {
            int col = nt * 16 + col0;
            float bc = bias[col];
#pragma unroll
            for (int v2 = 0; v2 < 8; ++v2) {
                if (mks[v2] != 0.0f) {
                    float y = eluf(acc[nt][v2] + bc);
                    atomicAdd(&outacc[(size_t)dn[v2] * 128 + col], (y - mean) * inv);
                }
            }
        }
    }
}

// ---------------- node-level kernels ----------------

// in-place: h = elu(acc / max(cnt,1))  (cnt may be null -> no division)
__global__ void node_finish_kernel(float* h, int total, const int* cnt) {
    int i = blockIdx.x * blockDim.x + threadIdx.x;
    if (i < total) {
        float v = h[i];
        if (cnt) {
            int c = cnt[i >> 7];
            v /= (float)(c > 0 ? c : 1);
        }
        h[i] = eluf(v);
    }
}

__global__ void degree_kernel(const int* __restrict__ dst, int* cnt) {
    int e = blockIdx.x * blockDim.x + threadIdx.x;
    if (e < cE) atomicAdd(&cnt[dst[e]], 1);
}

// global mean pool: one block per graph, 128 threads (one per channel)
__global__ void gmp_kernel(const float* __restrict__ h, float* xs, int npg, float inv) {
    int g = blockIdx.x, c = threadIdx.x;
    const float* b = h + (size_t)g * npg * 128;
    float s = 0.0f;
    for (int i = 0; i < npg; ++i) s += b[i * 128 + c];
    xs[g * 128 + c] = s * inv;
}

__global__ void score_pq_kernel(const float* __restrict__ h2, const float* __restrict__ wrel,
                                const float* __restrict__ wroot, float* p, float* q) {
    int n = blockIdx.x * blockDim.x + threadIdx.x;
    if (n < cN) {
        const float* row = h2 + (size_t)n * 128;
        float a = 0.0f, b = 0.0f;
        for (int k = 0; k < 128; ++k) {
            float v = row[k];
            a += v * wrel[k];
            b += v * wroot[k];
        }
        p[n] = a; q[n] = b;
    }
}

__global__ void edge_score_kernel(const int* __restrict__ src, const int* __restrict__ dst,
                                  const float* __restrict__ p, float* sacc) {
    int e = blockIdx.x * blockDim.x + threadIdx.x;
    if (e < cE) atomicAdd(&sacc[dst[e]], p[src[e]]);
}

__global__ void score_final_kernel(const float* __restrict__ q, const float* __restrict__ sacc,
                                   const float* __restrict__ brel, float* score) {
    int n = blockIdx.x * blockDim.x + threadIdx.x;
    if (n < cN) score[n] = tanhf(q[n] + brel[0] + sacc[n]);
}

// ---------------- top-K via per-graph bitonic sort (descending) ----------------

__global__ void __launch_bounds__(1024)
topk_kernel(const float* __restrict__ score, float* vals, int* perm, int* nmap) {
    __shared__ float sv[1024];
    __shared__ int   si[1024];
    int g = blockIdx.x, t = threadIdx.x;
    sv[t] = score[g * 1024 + t];
    si[t] = t;
    __syncthreads();
    for (int k = 2; k <= 1024; k <<= 1) {
        for (int j = k >> 1; j > 0; j >>= 1) {
            int ixj = t ^ j;
            if (ixj > t) {
                bool desc = ((t & k) == 0);
                float av = sv[t], bv = sv[ixj];
                bool sw = desc ? (av < bv) : (av > bv);
                if (sw) {
                    sv[t] = bv; sv[ixj] = av;
                    int ti = si[t]; si[t] = si[ixj]; si[ixj] = ti;
                }
            }
            __syncthreads();
        }
    }
    if (t < cK) {
        int orig = g * 1024 + si[t];
        perm[g * cK + t] = orig;
        vals[g * cK + t] = sv[t];
        nmap[orig] = g * cK + t;
    }
}

__global__ void x2_kernel(const float* __restrict__ h2, const int* __restrict__ perm,
                          const float* __restrict__ vals, float* x2) {
    int idx = blockIdx.x * blockDim.x + threadIdx.x;
    if (idx < cN2 * 128) {
        int i = idx >> 7, c = idx & 127;
        x2[idx] = h2[(size_t)perm[i] * 128 + c] * vals[i];
    }
}

// build remapped edge arrays + masks + per-node counts + total valid-edge count
__global__ void pool_edges_kernel(const int* __restrict__ src, const int* __restrict__ dst,
                                  const int* __restrict__ nmap, int* src2, int* dst2,
                                  float* emask, int* cnt3, int* vcnt) {
    __shared__ int csh;
    if (threadIdx.x == 0) csh = 0;
    __syncthreads();
    int e = blockIdx.x * blockDim.x + threadIdx.x;
    int ns = nmap[src[e]], nd = nmap[dst[e]];
    bool valid = (ns < cN2) && (nd < cN2);
    src2[e] = ns < (cN2 - 1) ? ns : (cN2 - 1);
    dst2[e] = nd < (cN2 - 1) ? nd : (cN2 - 1);
    emask[e] = valid ? 1.0f : 0.0f;
    if (valid) {
        atomicAdd(&cnt3[nd], 1);
        atomicAdd(&csh, 1);
    }
    __syncthreads();
    if (threadIdx.x == 0 && csh) atomicAdd(vcnt, csh);
}

// ---------------- classification head ----------------

__global__ void __launch_bounds__(128)
head_kernel(const float* __restrict__ xs0, const float* __restrict__ xs1,
            const float* __restrict__ xs2, const float* __restrict__ Wl1,
            const float* __restrict__ bl1, const float* __restrict__ Wl2,
            const float* __restrict__ bl2, float* out) {
    __shared__ float xf[384];
    __shared__ float z[128];
    __shared__ float lg[10];
    __shared__ float mred, lse;
    int g = blockIdx.x, t = threadIdx.x;
    xf[t]       = xs0[g * 128 + t];
    xf[128 + t] = xs1[g * 128 + t];
    xf[256 + t] = xs2[g * 128 + t];
    __syncthreads();
    float a = bl1[t];
    for (int i = 0; i < 384; ++i) a += xf[i] * Wl1[i * 128 + t];
    z[t] = eluf(a);
    __syncthreads();
    if (t < cNCLS) {
        float l = bl2[t];
        for (int k = 0; k < 128; ++k) l += z[k] * Wl2[k * cNCLS + t];
        lg[t] = l;
    }
    __syncthreads();
    if (t == 0) {
        float m = lg[0];
        for (int i = 1; i < cNCLS; ++i) m = fmaxf(m, lg[i]);
        float s = 0.0f;
        for (int i = 0; i < cNCLS; ++i) s += expf(lg[i] - m);
        mred = m; lse = logf(s);
    }
    __syncthreads();
    if (t < cNCLS) out[g * cNCLS + t] = lg[t] - mred - lse;
}

// ---------------- host launch ----------------

// workspace layout (float units)
static constexpr size_t OFF_H1    = 0;                           // N*128
static constexpr size_t OFF_H2    = OFF_H1 + (size_t)cN * 128;
static constexpr size_t OFF_H3    = OFF_H2 + (size_t)cN * 128;   // N2*128
static constexpr size_t OFF_X2    = OFF_H3 + (size_t)cN2 * 128;
static constexpr size_t OFF_P     = OFF_X2 + (size_t)cN2 * 128;  // N
static constexpr size_t OFF_Q     = OFF_P + cN;
static constexpr size_t OFF_SACC  = OFF_Q + cN;
static constexpr size_t OFF_SCORE = OFF_SACC + cN;
static constexpr size_t OFF_VALS  = OFF_SCORE + cN;              // N2
static constexpr size_t OFF_EMASK = OFF_VALS + cN2;              // E
static constexpr size_t OFF_STATS = OFF_EMASK + cE;              // 32
static constexpr size_t OFF_XS0   = OFF_STATS + 32;
static constexpr size_t OFF_XS1   = OFF_XS0 + (size_t)cG * 128;
static constexpr size_t OFF_XS2   = OFF_XS1 + (size_t)cG * 128;
static constexpr size_t OFF_PERM  = OFF_XS2 + (size_t)cG * 128;  // int N2
static constexpr size_t OFF_NMAP  = OFF_PERM + cN2;              // int N
static constexpr size_t OFF_CNT2  = OFF_NMAP + cN;               // int N
static constexpr size_t OFF_CNT3  = OFF_CNT2 + cN;               // int N2
static constexpr size_t OFF_SRC2  = OFF_CNT3 + cN2;              // int E
static constexpr size_t OFF_DST2  = OFF_SRC2 + cE;               // int E
static constexpr size_t OFF_VCNT  = OFF_DST2 + cE;               // int (pad 8)
static constexpr size_t OFF_W1BP  = OFF_VCNT + 8;                // 16384 bf16 -> 8192 floats
static constexpr size_t OFF_WC0P  = OFF_W1BP + 8192;             // 32768 bf16 -> 16384 floats
static constexpr size_t OFF_WC1P  = OFF_WC0P + 16384;

extern "C" void kernel_launch(void* const* d_in, const int* in_sizes, int n_in,
                              void* d_out, int out_size, void* d_ws, size_t ws_size,
                              hipStream_t stream) {
    const float* x    = (const float*)d_in[0];
    const int*   eidx = (const int*)d_in[1];
    const int*   src  = eidx;
    const int*   dst  = eidx + cE;
    const float* W1a  = (const float*)d_in[3];
    const float* b1a  = (const float*)d_in[4];
    const float* W1b  = (const float*)d_in[5];
    const float* b1b  = (const float*)d_in[6];
    const float* Wc0  = (const float*)d_in[7];
    const float* bc0  = (const float*)d_in[8];
    const float* Wc1  = (const float*)d_in[9];
    const float* bc1  = (const float*)d_in[10];
    const float* Wrel = (const float*)d_in[11];
    const float* brel = (const float*)d_in[12];
    const float* Wroot= (const float*)d_in[13];
    const float* Wl1  = (const float*)d_in[14];
    const float* bl1  = (const float*)d_in[15];
    const float* Wl2  = (const float*)d_in[16];
    const float* bl2  = (const float*)d_in[17];

    float* ws = (float*)d_ws;
    float*  H1    = ws + OFF_H1;
    float*  H2    = ws + OFF_H2;
    float*  H3    = ws + OFF_H3;
    float*  X2    = ws + OFF_X2;
    float*  P     = ws + OFF_P;
    float*  Q     = ws + OFF_Q;
    float*  SACC  = ws + OFF_SACC;
    float*  SCORE = ws + OFF_SCORE;
    float*  VALS  = ws + OFF_VALS;
    float*  EMASK = ws + OFF_EMASK;
    float*  STATS = ws + OFF_STATS;
    float*  XS0   = ws + OFF_XS0;
    float*  XS1   = ws + OFF_XS1;
    float*  XS2   = ws + OFF_XS2;
    int*    PERM  = (int*)(ws + OFF_PERM);
    int*    NMAP  = (int*)(ws + OFF_NMAP);
    int*    CNT2  = (int*)(ws + OFF_CNT2);
    int*    CNT3  = (int*)(ws + OFF_CNT3);
    int*    SRC2  = (int*)(ws + OFF_SRC2);
    int*    DST2  = (int*)(ws + OFF_DST2);
    int*    VCNT  = (int*)(ws + OFF_VCNT);
    bf16_t* W1BP  = (bf16_t*)(ws + OFF_W1BP);
    bf16_t* WC0P  = (bf16_t*)(ws + OFF_WC0P);
    bf16_t* WC1P  = (bf16_t*)(ws + OFF_WC1P);

    const int NH = cN * 128;    // 4194304
    const int N2H = cN2 * 128;  // 3358720
    const float totE = (float)cE * 128.0f;

    // --- init accumulators ---
    zerof_kernel<<<(NH + 255) / 256, 256, 0, stream>>>(H1, NH);
    zerof_kernel<<<(NH + 255) / 256, 256, 0, stream>>>(H2, NH);
    zerof_kernel<<<(N2H + 255) / 256, 256, 0, stream>>>(H3, N2H);
    zerof_kernel<<<(cN + 255) / 256, 256, 0, stream>>>(SACC, cN);
    zerof_kernel<<<1, 32, 0, stream>>>(STATS, 32);
    filli_kernel<<<(cN + 255) / 256, 256, 0, stream>>>(CNT2, cN, 0);
    filli_kernel<<<(cN2 + 255) / 256, 256, 0, stream>>>(CNT3, cN2, 0);
    filli_kernel<<<1, 32, 0, stream>>>(VCNT, 1, 0);
    filli_kernel<<<(cN + 255) / 256, 256, 0, stream>>>(NMAP, cN, cN2);

    // --- pack weights into WMMA B-fragment bf16 layout ---
    pack_w_kernel<<<(8 * 4 * 512) / 256, 256, 0, stream>>>(W1b, W1BP, 4);
    pack_w_kernel<<<(8 * 8 * 512) / 256, 256, 0, stream>>>(Wc0, WC0P, 8);
    pack_w_kernel<<<(8 * 8 * 512) / 256, 256, 0, stream>>>(Wc1, WC1P, 8);

    // --- EdgeConv 1 (two layers, global LN each, sum-aggregate) ---
    conv1_statsA_kernel<<<cE / 256, 256, 0, stream>>>(x, src, dst, W1a, b1a, STATS);
    finalize_stats_kernel<<<1, 32, 0, stream>>>(STATS, 0, totE, nullptr);
    conv1_wmma_kernel<<<cE / 128, 256, 0, stream>>>(x, src, dst, W1a, b1a, W1BP, b1b,
                                                    STATS, 0, nullptr);
    finalize_stats_kernel<<<1, 32, 0, stream>>>(STATS, 1, totE, nullptr);
    conv1_wmma_kernel<<<cE / 128, 256, 0, stream>>>(x, src, dst, W1a, b1a, W1BP, b1b,
                                                    STATS, 1, H1);
    node_finish_kernel<<<(NH + 255) / 256, 256, 0, stream>>>(H1, NH, nullptr);
    degree_kernel<<<cE / 256, 256, 0, stream>>>(dst, CNT2);
    gmp_kernel<<<cG, 128, 0, stream>>>(H1, XS0, cNPER, 1.0f / (float)cNPER);

    // --- EdgeConv 2 (K=256, mean-aggregate) ---
    edgeconv_wmma256_kernel<<<cE / 128, 256, 0, stream>>>(H1, src, dst, nullptr, WC0P, bc0,
                                                          STATS, 2, 0, nullptr, nullptr);
    finalize_stats_kernel<<<1, 32, 0, stream>>>(STATS, 2, totE, nullptr);
    edgeconv_wmma256_kernel<<<cE / 128, 256, 0, stream>>>(H1, src, dst, nullptr, WC0P, bc0,
                                                          STATS, 2, 1, H2, dst);
    node_finish_kernel<<<(NH + 255) / 256, 256, 0, stream>>>(H2, NH, CNT2);
    gmp_kernel<<<cG, 128, 0, stream>>>(H2, XS1, cNPER, 1.0f / (float)cNPER);

    // --- SAG pooling: score, top-K, gate ---
    score_pq_kernel<<<cN / 256, 256, 0, stream>>>(H2, Wrel, Wroot, P, Q);
    edge_score_kernel<<<cE / 256, 256, 0, stream>>>(src, dst, P, SACC);
    score_final_kernel<<<cN / 256, 256, 0, stream>>>(Q, SACC, brel, SCORE);
    topk_kernel<<<cG, 1024, 0, stream>>>(SCORE, VALS, PERM, NMAP);
    x2_kernel<<<(N2H + 255) / 256, 256, 0, stream>>>(H2, PERM, VALS, X2);
    pool_edges_kernel<<<cE / 256, 256, 0, stream>>>(src, dst, NMAP, SRC2, DST2, EMASK,
                                                    CNT3, VCNT);

    // --- EdgeConv 3 on pooled graph (masked LN, mean-aggregate) ---
    edgeconv_wmma256_kernel<<<cE / 128, 256, 0, stream>>>(X2, SRC2, DST2, EMASK, WC1P, bc1,
                                                          STATS, 3, 0, nullptr, nullptr);
    finalize_stats_kernel<<<1, 32, 0, stream>>>(STATS, 3, 0.0f, VCNT);
    edgeconv_wmma256_kernel<<<cE / 128, 256, 0, stream>>>(X2, SRC2, DST2, EMASK, WC1P, bc1,
                                                          STATS, 3, 1, H3, DST2);
    node_finish_kernel<<<(N2H + 255) / 256, 256, 0, stream>>>(H3, N2H, CNT3);
    gmp_kernel<<<cG, 128, 0, stream>>>(H3, XS2, cK, 1.0f / (float)cK);

    // --- head ---
    head_kernel<<<cG, 128, 0, stream>>>(XS0, XS1, XS2, Wl1, bl1, Wl2, bl2, (float*)d_out);
}
